// MultiHeadLatentAttention_69483980914988
// MI455X (gfx1250) — matrix operand order repaired and checked by reading, not compiled, not run on hardware
//
#include <hip/hip_runtime.h>

// MLA forward for MI455X (gfx1250): bf16 WMMA everywhere, fp32 accumulate.
// GEMMs stage tiles through LDS with double-buffered async global->LDS copies
// (global_load_async_to_lds_b128 / s_wait_asynccnt), feeding
// v_wmma_f32_16x16x32_bf16. Attention is flash-style with online softmax.

#define D_MODEL 5120
#define NUM_HEADS 128
#define HEAD_DIM 40
#define LATENT 1536
#define BATCH 2
#define SEQ 1024
#define M_TOTAL (BATCH * SEQ)  // 2048
#define DH_PAD 64
#define DH_PAD_V 48
#define SOFTMAX_SCALE 0.15811388300841898f  // 1/sqrt(40)

#define BLK_M 128
#define BLK_N 128
#define KSTEP 32
#define LPAD 40  // padded LDS row length (elements): 80B stride, conflict-free

typedef __bf16 bf16;
typedef __attribute__((ext_vector_type(8)))  bf16  v8bf;
typedef __attribute__((ext_vector_type(16))) bf16  v16bf;
typedef __attribute__((ext_vector_type(8)))  float v8f;

union BF16Frag { v16bf v; v8bf h[2]; };

__device__ __attribute__((always_inline)) inline v8f
wmma_bf16(v16bf a, v16bf b, v8f c) {
  return __builtin_amdgcn_wmma_f32_16x16x32_bf16(false, a, false, b,
                                                 (short)0, c, false, false);
}

// A-fragment (16x32 bf16): lane<16 holds K[0..7],K[16..23]; lane>=16: K[8..15],K[24..31].
__device__ __attribute__((always_inline)) inline v16bf
load_a_frag(const bf16* row_base, int lhalf) {
  BF16Frag u;
  const v8bf* p = (const v8bf*)(row_base + lhalf * 8);
  u.h[0] = p[0];
  u.h[1] = p[2];
  return u.v;
}

// B-fragment (32x16 bf16): lane<16 holds K[0..7],K[16..23]; lane>=16: K[8..15],K[24..31].
__device__ __attribute__((always_inline)) inline v16bf
load_b_frag(const bf16* col_base, int lhalf) {
  BF16Frag u;
  const v8bf* p = (const v8bf*)(col_base + lhalf * 8);
  u.h[0] = p[0];
  u.h[1] = p[2];
  return u.v;
}

// ---- CDNA5 async global->LDS (ASYNCcnt-tracked). INST_OFFSET adds to both
// the global and the LDS address, which we exploit for the +16B chunk.
__device__ __attribute__((always_inline)) inline void
async_b128(unsigned lds, const void* g) {
  asm volatile("global_load_async_to_lds_b128 %0, %1, off"
               :: "v"(lds), "v"(g) : "memory");
}
__device__ __attribute__((always_inline)) inline void
async_b128_off16(unsigned lds, const void* g) {
  asm volatile("global_load_async_to_lds_b128 %0, %1, off offset:16"
               :: "v"(lds), "v"(g) : "memory");
}
__device__ __attribute__((always_inline)) inline void wait_async_le4() {
  asm volatile("s_wait_asynccnt 0x4" ::: "memory");
}
__device__ __attribute__((always_inline)) inline void wait_async_0() {
  asm volatile("s_wait_asynccnt 0x0" ::: "memory");
}
// Flat->LDS aperture mapping keeps the wave-relative LDS byte offset in the
// low 32 bits of the flat address.
__device__ __attribute__((always_inline)) inline unsigned lds_off(const void* p) {
  return (unsigned)(unsigned long long)p;
}

// ---------------------------------------------------------------- pre-passes
__global__ __launch_bounds__(256) void cvt_f32_bf16(const float* __restrict__ src,
                                                    bf16* __restrict__ dst, long n) {
  long i = (long)blockIdx.x * blockDim.x + threadIdx.x;
  if (i < n) dst[i] = (bf16)src[i];
}

// W is [K,N] row-major f32; WT becomes [N,K] row-major bf16.
__global__ __launch_bounds__(256) void transpose_f32_bf16(const float* __restrict__ W,
                                                          bf16* __restrict__ WT,
                                                          int K, int N) {
  long i = (long)blockIdx.x * blockDim.x + threadIdx.x;
  long total = (long)K * N;
  if (i < total) {
    int k = (int)(i / N);
    int n = (int)(i % N);
    WT[(long)n * K + k] = (bf16)W[i];
  }
}

__global__ __launch_bounds__(256) void fill_zero_bf16(bf16* __restrict__ p, long n) {
  long i = (long)blockIdx.x * blockDim.x + threadIdx.x;
  if (i < n) p[i] = (bf16)0.0f;
}

// ---------------------------------------------------------------- GEMM
// C[M,N] = A[M,K] * BT[N,K]^T + bias. Block tile 128x128 (8 waves; each wave
// 32x64 = 2x4 WMMA tiles). Double-buffered async global->LDS staging.
// mode: 0 plain bf16 [M,N]; 1 q/k head layout; 2 v^T layout.
__global__ __launch_bounds__(256) void gemm_bf16_kernel(
    const bf16* __restrict__ A, const bf16* __restrict__ BT,
    const float* __restrict__ bias, bf16* __restrict__ out,
    int M, int N, int K, int mode) {
  __shared__ __align__(16) bf16 sA[2][BLK_M][LPAD];
  __shared__ __align__(16) bf16 sB[2][BLK_N][LPAD];

  const int tid = threadIdx.x;
  const int wave = tid >> 5, lane = tid & 31;
  const int lhalf = lane >> 4, lidx = lane & 15;
  const int waveM = wave & 3, waveN = wave >> 2;
  const int m0 = blockIdx.y * BLK_M;
  const int n0 = blockIdx.x * BLK_N;

  // Async-copy assignment: thread t fills LDS row t>>1, 32B at column (t&1)*16.
  const int ldRow = tid >> 1;
  const int ldCol = (tid & 1) * 16;  // element offset inside the 32-wide row
  const bf16* gA = A + (size_t)(m0 + ldRow) * K + ldCol;
  const bf16* gB = BT + (size_t)(n0 + ldRow) * K + ldCol;
  const unsigned ldsA[2] = {lds_off(&sA[0][ldRow][ldCol]), lds_off(&sA[1][ldRow][ldCol])};
  const unsigned ldsB[2] = {lds_off(&sB[0][ldRow][ldCol]), lds_off(&sB[1][ldRow][ldCol])};

  v8f acc[2][4] = {};

  // Prologue: fill buffer 0.
  async_b128(ldsA[0], gA);
  async_b128_off16(ldsA[0], gA);
  async_b128(ldsB[0], gB);
  async_b128_off16(ldsB[0], gB);

  const int nk = K / KSTEP;
  for (int ik = 0; ik < nk; ++ik) {
    const int cur = ik & 1;
    if (ik + 1 < nk) {
      const bf16* gAn = gA + (size_t)(ik + 1) * KSTEP;
      const bf16* gBn = gB + (size_t)(ik + 1) * KSTEP;
      async_b128(ldsA[1 - cur], gAn);
      async_b128_off16(ldsA[1 - cur], gAn);
      async_b128(ldsB[1 - cur], gBn);
      async_b128_off16(ldsB[1 - cur], gBn);
      wait_async_le4();  // the 4 older copies (buffer `cur`) have landed
    } else {
      wait_async_0();
    }
    __syncthreads();

    v16bf af[2], bfg[4];
#pragma unroll
    for (int a = 0; a < 2; ++a)
      af[a] = load_a_frag(&sA[cur][waveM * 32 + a * 16 + lidx][0], lhalf);
#pragma unroll
    for (int b = 0; b < 4; ++b)
      bfg[b] = load_b_frag(&sB[cur][waveN * 64 + b * 16 + lidx][0], lhalf);

#pragma unroll
    for (int a = 0; a < 2; ++a)
#pragma unroll
      for (int b = 0; b < 4; ++b)
        acc[a][b] = wmma_bf16(af[a], bfg[b], acc[a][b]);

    __syncthreads();  // reads done before next iteration's async overwrites
  }

#pragma unroll
  for (int a = 0; a < 2; ++a) {
#pragma unroll
    for (int b = 0; b < 4; ++b) {
      const int col = n0 + waveN * 64 + b * 16 + lidx;
      const float bval = bias[col];
#pragma unroll
      for (int r = 0; r < 8; ++r) {
        const int row = m0 + waveM * 32 + a * 16 + lhalf * 8 + r;
        const float val = acc[a][b][r] + bval;
        if (mode == 0) {
          out[(size_t)row * N + col] = (bf16)val;
        } else {
          const int batch = row >> 10, s = row & 1023;
          const int h = col / HEAD_DIM, dh = col % HEAD_DIM;
          const int bh = batch * NUM_HEADS + h;
          if (mode == 1)
            out[((size_t)bh * SEQ + s) * DH_PAD + dh] = (bf16)val;
          else
            out[((size_t)bh * DH_PAD_V + dh) * SEQ + s] = (bf16)val;
        }
      }
    }
  }
}

// ---------------------------------------------------------------- attention
// grid(SEQ/128, B*H), 256 threads = 8 waves; each wave owns 16 queries and
// streams key chunks of 32 with online softmax; O accumulator is 16x48 f32.
__global__ __launch_bounds__(256) void mla_attn_kernel(
    const bf16* __restrict__ qh, const bf16* __restrict__ kh,
    const bf16* __restrict__ vt, float* __restrict__ out) {
  __shared__ __align__(16) float sS[8][16][32];
  __shared__ __align__(16) bf16 sP[8][16][32];
  __shared__ float sFac[8][16];
  __shared__ float sL[8][16];

  const int tid = threadIdx.x;
  const int wave = tid >> 5, lane = tid & 31;
  const int lhalf = lane >> 4, lidx = lane & 15;
  const int bh = blockIdx.y;
  const int q0 = blockIdx.x * 128 + wave * 16;

  const bf16* Q = qh + (size_t)bh * SEQ * DH_PAD;
  const bf16* Kp = kh + (size_t)bh * SEQ * DH_PAD;
  const bf16* Vp = vt + (size_t)bh * DH_PAD_V * SEQ;

  v16bf qa0, qa1;
  {
    const bf16* qrow = Q + (size_t)(q0 + lidx) * DH_PAD;
    qa0 = load_a_frag(qrow, lhalf);
    qa1 = load_a_frag(qrow + 32, lhalf);
  }

  v8f o[3] = {};
  float run_m = -1e30f, run_l = 0.0f;  // valid in lanes < 16 (row = lane)

  for (int kc = 0; kc < SEQ / 32; ++kc) {
    const int kbase = kc * 32;

#pragma unroll
    for (int t = 0; t < 2; ++t) {
      const bf16* krow = Kp + (size_t)(kbase + t * 16 + lidx) * DH_PAD;
      v16bf kb0 = load_b_frag(krow, lhalf);
      v16bf kb1 = load_b_frag(krow + 32, lhalf);
      v8f s = {};
      s = wmma_bf16(qa0, kb0, s);
      s = wmma_bf16(qa1, kb1, s);
#pragma unroll
      for (int r = 0; r < 8; ++r)
        sS[wave][lhalf * 8 + r][t * 16 + lidx] = s[r] * SOFTMAX_SCALE;
    }
    __syncthreads();

    if (lane < 16) {
      float buf[32];
      float cmax = -1e30f;
#pragma unroll
      for (int c = 0; c < 32; ++c) {
        buf[c] = sS[wave][lane][c];
        cmax = fmaxf(cmax, buf[c]);
      }
      const float nm = fmaxf(run_m, cmax);
      const float fac = __expf(run_m - nm);
      float ls = run_l * fac;
#pragma unroll
      for (int c = 0; c < 32; ++c) {
        const float p = __expf(buf[c] - nm);
        ls += p;
        sP[wave][lane][c] = (bf16)p;
      }
      run_m = nm;
      run_l = ls;
      sFac[wave][lane] = fac;
    }
    __syncthreads();

    {
      float f[8];
#pragma unroll
      for (int r = 0; r < 8; ++r) f[r] = sFac[wave][lhalf * 8 + r];
#pragma unroll
      for (int nt = 0; nt < 3; ++nt)
#pragma unroll
        for (int r = 0; r < 8; ++r) o[nt][r] *= f[r];
    }

    v16bf pf;
    {
      BF16Frag u;
      const v8bf* p = (const v8bf*)(&sP[wave][lidx][lhalf * 8]);
      u.h[0] = p[0];
      u.h[1] = p[2];
      pf = u.v;
    }

#pragma unroll
    for (int nt = 0; nt < 3; ++nt) {
      const bf16* vrow = Vp + (size_t)(nt * 16 + lidx) * SEQ + kbase;
      v16bf vb = load_b_frag(vrow, lhalf);
      o[nt] = wmma_bf16(pf, vb, o[nt]);
    }
    __syncthreads();
  }

  if (lane < 16) sL[wave][lane] = run_l;
  __syncthreads();

  const int b = bh >> 7, h = bh & (NUM_HEADS - 1);
  float linv[8];
#pragma unroll
  for (int r = 0; r < 8; ++r) linv[r] = 1.0f / sL[wave][lhalf * 8 + r];

#pragma unroll
  for (int nt = 0; nt < 3; ++nt) {
    const int dh = nt * 16 + lidx;
    if (dh < HEAD_DIM) {
#pragma unroll
      for (int r = 0; r < 8; ++r) {
        const int q = q0 + lhalf * 8 + r;
        out[((size_t)(b * SEQ + q)) * D_MODEL + h * HEAD_DIM + dh] =
            o[nt][r] * linv[r];
      }
    }
  }
}

// ---------------------------------------------------------------- launcher
extern "C" void kernel_launch(void* const* d_in, const int* in_sizes, int n_in,
                              void* d_out, int out_size, void* d_ws, size_t ws_size,
                              hipStream_t stream) {
  const float* x  = (const float*)d_in[0];
  const float* Wq = (const float*)d_in[1];
  const float* bq = (const float*)d_in[2];
  const float* Wc = (const float*)d_in[3];
  const float* bc = (const float*)d_in[4];
  const float* Wk = (const float*)d_in[5];
  const float* bk = (const float*)d_in[6];
  const float* Wv = (const float*)d_in[7];
  const float* bv = (const float*)d_in[8];
  float* out = (float*)d_out;

  char* w = (char*)d_ws;
  size_t off = 0;
  auto alloc = [&](size_t elems) -> bf16* {
    bf16* p = (bf16*)(w + off);
    off += ((elems * sizeof(bf16)) + 255) & ~(size_t)255;
    return p;
  };
  bf16* xb  = alloc((size_t)M_TOTAL * D_MODEL);
  bf16* kvb = alloc((size_t)M_TOTAL * LATENT);
  bf16* WqT = alloc((size_t)D_MODEL * D_MODEL);
  bf16* WcT = alloc((size_t)LATENT * D_MODEL);
  bf16* WkT = alloc((size_t)D_MODEL * LATENT);
  bf16* WvT = alloc((size_t)D_MODEL * LATENT);
  const size_t qkElems = (size_t)BATCH * NUM_HEADS * SEQ * DH_PAD;
  const size_t vtElems = (size_t)BATCH * NUM_HEADS * DH_PAD_V * SEQ;
  bf16* qhd = alloc(qkElems);
  bf16* khd = alloc(qkElems);
  bf16* vtd = alloc(vtElems);

  const dim3 blk(256);
  auto nblk = [](long n) { return dim3((unsigned)((n + 255) / 256)); };

  cvt_f32_bf16<<<nblk((long)M_TOTAL * D_MODEL), blk, 0, stream>>>(
      x, xb, (long)M_TOTAL * D_MODEL);
  transpose_f32_bf16<<<nblk((long)D_MODEL * D_MODEL), blk, 0, stream>>>(
      Wq, WqT, D_MODEL, D_MODEL);
  transpose_f32_bf16<<<nblk((long)D_MODEL * LATENT), blk, 0, stream>>>(
      Wc, WcT, D_MODEL, LATENT);
  transpose_f32_bf16<<<nblk((long)LATENT * D_MODEL), blk, 0, stream>>>(
      Wk, WkT, LATENT, D_MODEL);
  transpose_f32_bf16<<<nblk((long)LATENT * D_MODEL), blk, 0, stream>>>(
      Wv, WvT, LATENT, D_MODEL);
  fill_zero_bf16<<<nblk((long)qkElems), blk, 0, stream>>>(qhd, (long)qkElems);
  fill_zero_bf16<<<nblk((long)qkElems), blk, 0, stream>>>(khd, (long)qkElems);
  fill_zero_bf16<<<nblk((long)vtElems), blk, 0, stream>>>(vtd, (long)vtElems);

  // kv = xb @ Wc + bc  -> kvb (plain bf16)
  gemm_bf16_kernel<<<dim3(LATENT / BLK_N, M_TOTAL / BLK_M), blk, 0, stream>>>(
      xb, WcT, bc, kvb, M_TOTAL, LATENT, D_MODEL, 0);
  // q = xb @ Wq + bq   -> qh head-padded layout
  gemm_bf16_kernel<<<dim3(D_MODEL / BLK_N, M_TOTAL / BLK_M), blk, 0, stream>>>(
      xb, WqT, bq, qhd, M_TOTAL, D_MODEL, D_MODEL, 1);
  // k = kvb @ Wk + bk  -> kh head-padded layout
  gemm_bf16_kernel<<<dim3(D_MODEL / BLK_N, M_TOTAL / BLK_M), blk, 0, stream>>>(
      kvb, WkT, bk, khd, M_TOTAL, D_MODEL, LATENT, 1);
  // v = kvb @ Wv + bv  -> vt transposed layout
  gemm_bf16_kernel<<<dim3(D_MODEL / BLK_N, M_TOTAL / BLK_M), blk, 0, stream>>>(
      kvb, WvT, bv, vtd, M_TOTAL, D_MODEL, LATENT, 2);

  // attention
  mla_attn_kernel<<<dim3(SEQ / 128, BATCH * NUM_HEADS), blk, 0, stream>>>(
      qhd, khd, vtd, out);
}
